// BiRNNStack_16381005267476
// MI455X (gfx1250) — matrix operand order repaired and checked by reading, not compiled, not run on hardware
//
#include <hip/hip_runtime.h>

typedef __bf16 bf16;
typedef __attribute__((ext_vector_type(16))) __bf16 v16bf;
typedef __attribute__((ext_vector_type(8)))  __bf16 v8bf;
typedef __attribute__((ext_vector_type(8)))  float  v8f;
typedef __attribute__((ext_vector_type(4)))  float  v4f;

#define RNN_T 1024
#define RNN_B 32
#define RNN_I 1024
#define RNN_H 256
#define NOUT  512   // 2*H concat feature width

// ---------------------------------------------------------------- utilities

__device__ __forceinline__ float fast_tanh(float x) {
  // tanh(x) = sign(x) * (1 - e) / (1 + e),  e = exp(-2|x|) = 2^(-2|x|*log2 e)
  float ax = __builtin_fabsf(x);
  float e  = __builtin_amdgcn_exp2f(ax * -2.885390081777927f);
  float r  = (1.0f - e) * __builtin_amdgcn_rcpf(1.0f + e);
  return __builtin_copysignf(r, x);
}

__global__ void cvt_f32_to_bf16(const float* __restrict__ in,
                                bf16* __restrict__ out, int n) {
  int i = blockIdx.x * blockDim.x + threadIdx.x;
  if (i < n) out[i] = (bf16)in[i];
}

__global__ void combine_bias(const float* __restrict__ a,
                             const float* __restrict__ b,
                             float* __restrict__ out, int n) {
  int i = blockIdx.x * blockDim.x + threadIdx.x;
  if (i < n) out[i] = a[i] + b[i];
}

// ------------------------------------------------- input-projection GEMM
// out[m][n] = sum_k A[m][k] * W[n][k] + bias[n],  out has row stride NOUT.
// MODE 0: A is x with layout (B, T, K), m = t*B + b.
// MODE 1: A is flat [M][K] (K == NOUT).
// Block: 256 threads (8 waves), tile M=128 x N=64; each wave does 4 M-subtiles
// sharing one B fragment. A tile staged to LDS as bf16 (converted in-flight);
// W rows [n][k] ARE the per-lane WMMA B-fragment layout -> contiguous loads.
template <int MODE>
__global__ __launch_bounds__(256) void proj_gemm(
    const float* __restrict__ A, const bf16* __restrict__ W,
    const float* __restrict__ bias, float* __restrict__ out, int K) {
  __shared__ __align__(16) bf16 As[128][48];   // 32-wide k-slab, padded stride

  const int tid  = threadIdx.x;
  const int lane = tid & 31;
  const int w    = tid >> 5;
  const int mBlk = blockIdx.x * 128;
  const int n0   = blockIdx.y * 64 + (w & 3) * 16;
  const int mW   = (w >> 2) * 16;            // 0 or 16
  const int srow = tid >> 1;                 // staging row 0..127
  const int skk  = (tid & 1) * 16;           // staging k segment

  const int nn = n0 + (lane & 15);           // this lane's output column
  const int bk = (lane & 16);                // B-frag k half (0|16)
  const int ak = (lane & 16);                // A-frag k base
  const int ar = (lane & 15);                // A-frag row within tile

  const float bs = bias[nn];

  v8f zero = {};
  v8f acc[4] = {zero, zero, zero, zero};

  for (int kk = 0; kk < K; kk += 32) {
    // ---- stage 128x32 A slab into LDS as bf16 ----
    const float* p;
    if (MODE == 0) {
      int m = mBlk + srow;
      int b = m & (RNN_B - 1);
      int t = m >> 5;
      p = A + ((size_t)b * RNN_T + t) * (size_t)K + kk + skk;
    } else {
      p = A + (size_t)(mBlk + srow) * (size_t)K + kk + skk;
    }
    const v4f* pv = (const v4f*)p;
    v4f f0 = pv[0], f1 = pv[1], f2 = pv[2], f3 = pv[3];
    v8bf o0, o1;
#pragma unroll
    for (int j = 0; j < 4; ++j) {
      o0[j]     = (bf16)f0[j];
      o0[4 + j] = (bf16)f1[j];
      o1[j]     = (bf16)f2[j];
      o1[4 + j] = (bf16)f3[j];
    }
    *(v8bf*)&As[srow][skk]     = o0;
    *(v8bf*)&As[srow][skk + 8] = o1;
    __syncthreads();

    // ---- B fragment: contiguous 32B from global (weights live in L2) ----
    v16bf bf = *(const v16bf*)(W + (size_t)nn * K + kk + bk);

#pragma unroll
    for (int i = 0; i < 4; ++i) {
      int arow = mW + 32 * i + ar;
      v8bf lo = *(const v8bf*)&As[arow][ak];
      v8bf hi = *(const v8bf*)&As[arow][ak + 8];
      v16bf af = __builtin_shufflevector(lo, hi,
                   0,1,2,3,4,5,6,7,8,9,10,11,12,13,14,15);
      acc[i] = __builtin_amdgcn_wmma_f32_16x16x32_bf16(
                   false, af, false, bf, (short)0, acc[i], false, false);
    }
    __syncthreads();
  }

#pragma unroll
  for (int i = 0; i < 4; ++i) {
#pragma unroll
    for (int r = 0; r < 8; ++r) {
      int m = mBlk + mW + 32 * i + r + ((lane & 16) ? 8 : 0);
      out[(size_t)m * NOUT + nn] = acc[i][r] + bs;
    }
  }
}

// ------------------------------------------------------------- recurrence
// One persistent workgroup per direction (grid.x = 2), 512 threads = 16 waves.
// Wave w owns output columns n0 = w*16 and computes BOTH 16-row batch tiles
// (m0 = 0 and 16) against the same register-resident Whh B-fragments.
// 16 waves -> 4 waves/SIMD -> ~256 VGPR/wave budget, so bfr[8] (64 VGPRs)
// stays in registers for all T steps (no scratch spills on the serial path).
// Two independent WMMA chains per wave interleave in the matrix pipe.
// h is double-buffered bf16 in LDS -> exactly one barrier per timestep.
__global__ __launch_bounds__(512) void rnn_recurrence(
    const float* __restrict__ xp,     // [T*B][NOUT], this dir at col d*H
    const bf16* __restrict__ WhhB,    // [2][H][H] bf16
    float* __restrict__ out, int finalLayout) {
  const int d    = blockIdx.x;
  const int tid  = threadIdx.x;
  const int lane = tid & 31;
  const int w    = tid >> 5;          // 0..15
  const int n0   = w * 16;

  __shared__ __align__(16) bf16 hT[2][RNN_B][264];   // double-buffered h (bf16)

  for (int i = tid; i < 2 * RNN_B * 264; i += 512) ((bf16*)hT)[i] = (bf16)0.0f;
  __syncthreads();

  // persistent Whh B-fragments (64 VGPRs), reused every timestep
  const bf16* Wd = WhhB + (size_t)d * RNN_H * RNN_H;
  const int nn = n0 + (lane & 15);
  const int kb = (lane & 16);
  v16bf bfr[8];
#pragma unroll
  for (int ks = 0; ks < 8; ++ks)
    bfr[ks] = *(const v16bf*)(Wd + (size_t)nn * RNN_H + ks * 32 + kb);

  const int arA = (lane & 15);        // batch rows 0..15
  const int arB = 16 + (lane & 15);   // batch rows 16..31
  const int ak  = (lane & 16);
  const int mHi = (lane & 16) ? 8 : 0;

  int p = 0;
  for (int step = 0; step < RNN_T; ++step) {
    const int t = d ? (RNN_T - 1 - step) : step;
    const float* xr = xp + (size_t)t * RNN_B * NOUT + (size_t)d * RNN_H;

    // prefetch next step's xp slab behind the WMMA chain
    if (step + 1 < RNN_T) {
      const int tn = d ? (RNN_T - 2 - step) : step + 1;
      __builtin_prefetch(xp + (size_t)tn * RNN_B * NOUT + (size_t)d * RNN_H +
                             (size_t)(lane & 15) * NOUT, 0, 0);
    }

    // pull this step's xp early so its latency overlaps the WMMAs
    float xvA[8], xvB[8];
#pragma unroll
    for (int r = 0; r < 8; ++r) {
      xvA[r] = xr[(size_t)(r + mHi) * NOUT + nn];
      xvB[r] = xr[(size_t)(16 + r + mHi) * NOUT + nn];
    }

    v8f c0 = {}, c1 = {};
#pragma unroll
    for (int ks = 0; ks < 8; ++ks) {
      v8bf loA = *(const v8bf*)&hT[p][arA][ks * 32 + ak];
      v8bf hiA = *(const v8bf*)&hT[p][arA][ks * 32 + ak + 8];
      v8bf loB = *(const v8bf*)&hT[p][arB][ks * 32 + ak];
      v8bf hiB = *(const v8bf*)&hT[p][arB][ks * 32 + ak + 8];
      v16bf afA = __builtin_shufflevector(loA, hiA,
                    0,1,2,3,4,5,6,7,8,9,10,11,12,13,14,15);
      v16bf afB = __builtin_shufflevector(loB, hiB,
                    0,1,2,3,4,5,6,7,8,9,10,11,12,13,14,15);
      c0 = __builtin_amdgcn_wmma_f32_16x16x32_bf16(
               false, afA, false, bfr[ks], (short)0, c0, false, false);
      c1 = __builtin_amdgcn_wmma_f32_16x16x32_bf16(
               false, afB, false, bfr[ks], (short)0, c1, false, false);
    }

    const size_t col = (size_t)d * RNN_H + nn;
#pragma unroll
    for (int r = 0; r < 8; ++r) {
      const int mA = r + mHi;                     // batch rows 0..15
      const int mB = 16 + r + mHi;                // batch rows 16..31
      const float hA = fast_tanh(c0[r] + xvA[r]);
      const float hB = fast_tanh(c1[r] + xvB[r]);
      hT[1 - p][mA][nn] = (bf16)hA;
      hT[1 - p][mB][nn] = (bf16)hB;
      const size_t oA = finalLayout
          ? ((size_t)mA * RNN_T + t) * NOUT + col   // (B, T, 2H)
          : ((size_t)t * RNN_B + mA) * NOUT + col;  // (T*B, 2H) for layer 1
      const size_t oB = finalLayout
          ? ((size_t)mB * RNN_T + t) * NOUT + col
          : ((size_t)t * RNN_B + mB) * NOUT + col;
      out[oA] = hA;
      out[oB] = hB;
    }
    __syncthreads();   // new h visible before next step's fragment loads
    p ^= 1;
  }
}

// ---------------------------------------------------------------- launcher

extern "C" void kernel_launch(void* const* d_in, const int* in_sizes, int n_in,
                              void* d_out, int out_size, void* d_ws,
                              size_t ws_size, hipStream_t stream) {
  (void)in_sizes; (void)n_in; (void)out_size; (void)ws_size;

  const float* x    = (const float*)d_in[0];
  const float* Wih0 = (const float*)d_in[1];   // (2, H, I)
  const float* Whh0 = (const float*)d_in[2];   // (2, H, H)
  const float* bih0 = (const float*)d_in[3];   // (2, H)
  const float* bhh0 = (const float*)d_in[4];
  const float* Wih1 = (const float*)d_in[5];   // (2, H, 2H)
  const float* Whh1 = (const float*)d_in[6];
  const float* bih1 = (const float*)d_in[7];
  const float* bhh1 = (const float*)d_in[8];

  char* ws = (char*)d_ws;                      // needs ~132 MiB
  bf16*  Wih0b = (bf16*)(ws + 0x000000);       // 1 MiB
  bf16*  Whh0b = (bf16*)(ws + 0x100000);       // 256 KiB
  bf16*  Wih1b = (bf16*)(ws + 0x140000);       // 512 KiB
  bf16*  Whh1b = (bf16*)(ws + 0x1C0000);       // 256 KiB
  float* bias0 = (float*)(ws + 0x200000);      // 2 KiB
  float* bias1 = (float*)(ws + 0x201000);      // 2 KiB
  float* xp    = (float*)(ws + 0x400000);      // 64 MiB (reused for layer 1)
  float* h0    = (float*)(ws + 0x4400000);     // 64 MiB layer-0 output

  // weight pack (f32 -> bf16) + fused biases
  cvt_f32_to_bf16<<<2048, 256, 0, stream>>>(Wih0, Wih0b, 2 * RNN_H * RNN_I);
  cvt_f32_to_bf16<<<512,  256, 0, stream>>>(Whh0, Whh0b, 2 * RNN_H * RNN_H);
  cvt_f32_to_bf16<<<1024, 256, 0, stream>>>(Wih1, Wih1b, 2 * RNN_H * NOUT);
  cvt_f32_to_bf16<<<512,  256, 0, stream>>>(Whh1, Whh1b, 2 * RNN_H * RNN_H);
  combine_bias<<<2, 256, 0, stream>>>(bih0, bhh0, bias0, NOUT);
  combine_bias<<<2, 256, 0, stream>>>(bih1, bhh1, bias1, NOUT);

  dim3 pg(RNN_T * RNN_B / 128, NOUT / 64);     // (256, 8)

  // layer 0
  proj_gemm<0><<<pg, 256, 0, stream>>>(x, Wih0b, bias0, xp, RNN_I);
  rnn_recurrence<<<2, 512, 0, stream>>>(xp, Whh0b, h0, 0);
  // layer 1 (consumes concat(fwd,bwd) of layer 0)
  proj_gemm<1><<<pg, 256, 0, stream>>>(h0, Wih1b, bias1, xp, NOUT);
  rnn_recurrence<<<2, 512, 0, stream>>>(xp, Whh1b, (float*)d_out, 1);
}
